// CNFBlock_89086211653842
// MI455X (gfx1250) — compile-verified, matrix operand order checked by reading, not run on hardware
//
#include <hip/hip_runtime.h>
#include <hip/hip_bf16.h>

// CNF block for MI455X (gfx1250, wave32, WMMA).
// 160000 rows x E=128, 2 RK4 steps (8 f-evals), each eval = 2 GEMMs done with
// v_wmma_f32_16x16x32_f16 (f16 operands, f32 accum). State stays in LDS for
// the whole integration; only emb/log_pz0/out touch HBM per row.

typedef _Float16 h16;
typedef __attribute__((ext_vector_type(16))) _Float16 v16h;
typedef __attribute__((ext_vector_type(8)))  _Float16 v8h;
typedef __attribute__((ext_vector_type(8)))  float    v8f;

#define E_DIM 128
#define T_TOK 10000
#define NWAVE 4            // waves per workgroup
#define ROWS_W 16          // rows per wave (one 16x128 WMMA tile)

// Load a 16x32 f16 fragment in the CDNA5 A-layout (also used for B = A-layout
// of B^T) from an LDS matrix stored row-major [row][k], k contiguous.
// lanes 0-15:  row = lane,    elems 0..7 = k kb+0..7,   8..15 = kb+16..23
// lanes 16-31: row = lane-16, elems 0..7 = k kb+8..15,  8..15 = kb+24..31
__device__ __forceinline__ v16h load_frag(const h16* __restrict__ buf,
                                          int row, int kb, int half_off) {
  const int off = row * E_DIM + kb + half_off;      // multiple of 8 -> 16B aligned
  v8h lo = *(const v8h*)(buf + off);
  v8h hi = *(const v8h*)(buf + off + 16);
  v16h r;
#pragma unroll
  for (int i = 0; i < 8; ++i) { r[i] = lo[i]; r[i + 8] = hi[i]; }
  return r;
}

__global__ void __launch_bounds__(NWAVE * 32)
cnf_rk4_kernel(const float* __restrict__ h_in,   // [16,128]
               const float* __restrict__ emb,    // [10000,128]
               const float* __restrict__ lpz0,   // [16*10000]
               const float* __restrict__ Wx,     // [128,128]
               const float* __restrict__ wxt,    // [128]
               const float* __restrict__ bx,     // [128]
               const float* __restrict__ Wh,     // [128,128]
               const float* __restrict__ wht,    // [128]
               const float* __restrict__ bh,     // [128]
               const float* __restrict__ W2,     // [128,128]
               const float* __restrict__ b2,     // [128]
               float* __restrict__ out)          // [160000]
{
  extern __shared__ char smem[];
  h16*   sWx     = (h16*)smem;                               // 128*128 f16
  h16*   sW2     = sWx + E_DIM * E_DIM;                      // 128*128 f16
  float* sc      = (float*)(sW2 + E_DIM * E_DIM);            // 128 f32
  float* stw     = sc + E_DIM;                               // 128 f32
  float* sb2     = stw + E_DIM;                              // 128 f32
  float* szb_all = sb2 + E_DIM;                              // NWAVE*16*128 f32
  h16*   sst_all = (h16*)(szb_all + NWAVE * ROWS_W * E_DIM); // NWAVE*16*128 f16
  float* shb_all = (float*)(sst_all + NWAVE * ROWS_W * E_DIM); // NWAVE*128 f32

  const int tid  = threadIdx.x;
  const int lane = tid & 31;
  const int wave = tid >> 5;
  const int nthr = NWAVE * 32;

  // ---- workgroup init: weights -> f16 LDS, c = einsum('ij,ji->j',W2,Wx) ----
  for (int i = tid; i < E_DIM * E_DIM; i += nthr) {
    sWx[i] = (h16)Wx[i];
    sW2[i] = (h16)W2[i];
  }
  if (tid < E_DIM) {
    float acc = 0.f;
    for (int i = 0; i < E_DIM; ++i) acc += W2[i * E_DIM + tid] * Wx[tid * E_DIM + i];
    sc[tid]  = acc;
    stw[tid] = wxt[tid] + wht[tid];
    sb2[tid] = b2[tid];
  }

  // ---- per-wave init: row tile, hb vector, z0 state ----
  float* szb = szb_all + wave * ROWS_W * E_DIM;
  h16*   sst = sst_all + wave * ROWS_W * E_DIM;
  float* shb = shb_all + wave * E_DIM;

  const int wrow = blockIdx.x * NWAVE + wave;   // 0..9999
  const int base = wrow * ROWS_W;               // global row base
  const int sb   = base / T_TOK;                // single sb per wave (10000%16==0)
  const int t0   = base % T_TOK;

  for (int e = lane; e < E_DIM; e += 32) {      // hb = h[sb]@Wh^T + bh + bx
    float acc = bh[e] + bx[e];
    for (int k = 0; k < E_DIM; ++k) acc += h_in[sb * E_DIM + k] * Wh[e * E_DIM + k];
    shb[e] = acc;
  }
  for (int idx = lane; idx < ROWS_W * E_DIM; idx += 32) {   // z0 = emb rows
    int m = idx >> 7, e = idx & (E_DIM - 1);
    szb[idx] = emb[(t0 + m) * E_DIM + e];
  }
  __syncthreads();

  // D-layout helpers (v_wmma 16x16 f32 C/D: VGPR r -> M=r | r+8, lane -> N)
  const int half_off = (lane < 16) ? 0 : 8;
  const int l15 = lane & 15;
  const int mD  = (lane < 16) ? 0 : 8;

  const float dt = 0.5f;
  float divacc[8];
#pragma unroll
  for (int r = 0; r < 8; ++r) divacc[r] = 0.f;

  v8f dz[8];

  for (int step = 0; step < 2; ++step) {
    const float tb = (float)step * dt;
    v8f kacc[8];
#pragma unroll
    for (int nt = 0; nt < 8; ++nt) kacc[nt] = {};

    for (int s = 0; s < 4; ++s) {
      const float a    = (s == 0) ? 0.f : ((s == 3) ? dt : 0.5f * dt);
      const float w    = (s == 1 || s == 2) ? (dt / 3.f) : (dt / 6.f);
      const float tcur = tb + a;  // RK4 stage times coincide with stage coeffs

      // ---- build f16 stage state: zs = zb + a * dz_prev ----
      if (s == 0) {
        for (int idx = lane; idx < ROWS_W * E_DIM; idx += 32)
          sst[idx] = (h16)szb[idx];
      } else {
#pragma unroll
        for (int nt = 0; nt < 8; ++nt) {
          const int hcol = nt * 16 + l15;
#pragma unroll
          for (int r = 0; r < 8; ++r) {
            const int m = mD + r;
            sst[m * E_DIM + hcol] = (h16)(szb[m * E_DIM + hcol] + a * dz[nt][r]);
          }
        }
      }

      // ---- A fragments of stage state (same-wave LDS ops are in order) ----
      v16h af[4];
#pragma unroll
      for (int kc = 0; kc < 4; ++kc) af[kc] = load_frag(sst, l15, kc * 32, half_off);

      // ---- GEMM1: pre = zs @ Wx^T + hb + t*tw ; softplus -> LDS, sigma -> div
#pragma unroll
      for (int nt = 0; nt < 8; ++nt) {
        const int hcol = nt * 16 + l15;
        v8f acc = {};
#pragma unroll
        for (int kc = 0; kc < 4; ++kc) {
          v16h bf = load_frag(sWx, hcol, kc * 32, half_off);
          acc = __builtin_amdgcn_wmma_f32_16x16x32_f16(false, af[kc], false, bf,
                                                       (short)0, acc, false, false);
        }
        const float bias = shb[hcol] + tcur * stw[hcol];
        const float wdiv = w * sc[hcol];
#pragma unroll
        for (int r = 0; r < 8; ++r) {
          const float p  = acc[r] + bias;
          const float en = __expf(-fabsf(p));
          const float sg = (p >= 0.f) ? 1.f / (1.f + en) : en / (1.f + en);
          divacc[r] += wdiv * sg;                                // exact divergence
          const float sp = fmaxf(p, 0.f) + __logf(1.f + en);     // stable softplus
          sst[(mD + r) * E_DIM + hcol] = (h16)sp;                // D -> A via LDS
        }
      }

      // ---- GEMM2: dz = softplus(pre) @ W2^T + b2 ----
#pragma unroll
      for (int kc = 0; kc < 4; ++kc) af[kc] = load_frag(sst, l15, kc * 32, half_off);
#pragma unroll
      for (int nt = 0; nt < 8; ++nt) {
        const int ocol = nt * 16 + l15;
        v8f acc = {};
#pragma unroll
        for (int kc = 0; kc < 4; ++kc) {
          v16h bf = load_frag(sW2, ocol, kc * 32, half_off);
          acc = __builtin_amdgcn_wmma_f32_16x16x32_f16(false, af[kc], false, bf,
                                                       (short)0, acc, false, false);
        }
        const float bo = sb2[ocol];
#pragma unroll
        for (int r = 0; r < 8; ++r) {
          acc[r] += bo;
          kacc[nt][r] += w * acc[r];
        }
        dz[nt] = acc;
      }
    } // stages

    // ---- zb += dt/6 * (k1 + 2k2 + 2k3 + k4) ----
#pragma unroll
    for (int nt = 0; nt < 8; ++nt) {
      const int hcol = nt * 16 + l15;
#pragma unroll
      for (int r = 0; r < 8; ++r) {
        const int m = mD + r;
        szb[m * E_DIM + hcol] += kacc[nt][r];
      }
    }
  } // steps

  // ---- reduce divergence over the 16-lane halves, write log_pz1 ----
#pragma unroll
  for (int r = 0; r < 8; ++r) {
    float v = divacc[r];
    v += __shfl_xor(v, 1, 32);
    v += __shfl_xor(v, 2, 32);
    v += __shfl_xor(v, 4, 32);
    v += __shfl_xor(v, 8, 32);
    if (l15 == 0) {
      const int row = base + mD + r;
      out[row] = lpz0[row] - v;
    }
  }
}

extern "C" void kernel_launch(void* const* d_in, const int* in_sizes, int n_in,
                              void* d_out, int out_size, void* d_ws, size_t ws_size,
                              hipStream_t stream) {
  (void)in_sizes; (void)n_in; (void)out_size; (void)d_ws; (void)ws_size;
  const float* h_in = (const float*)d_in[0];
  const float* emb  = (const float*)d_in[1];
  const float* lpz  = (const float*)d_in[2];
  const float* Wx   = (const float*)d_in[3];
  const float* wxt  = (const float*)d_in[4];
  const float* bxv  = (const float*)d_in[5];
  const float* Wh   = (const float*)d_in[6];
  const float* wht  = (const float*)d_in[7];
  const float* bhv  = (const float*)d_in[8];
  const float* W2   = (const float*)d_in[9];
  const float* b2   = (const float*)d_in[10];
  float* out = (float*)d_out;

  const int total_rows = 16 * T_TOK;                       // 160000
  const int blocks = total_rows / (NWAVE * ROWS_W);        // 2500

  size_t shmem = (size_t)(2 * E_DIM * E_DIM) * sizeof(h16) // sWx, sW2
               + (size_t)(3 * E_DIM) * sizeof(float)       // c, tw, b2
               + (size_t)(NWAVE * ROWS_W * E_DIM) * sizeof(float) // zb
               + (size_t)(NWAVE * ROWS_W * E_DIM) * sizeof(h16)   // stage/sp
               + (size_t)(NWAVE * E_DIM) * sizeof(float);  // hb
  hipLaunchKernelGGL(cnf_rk4_kernel, dim3(blocks), dim3(NWAVE * 32), shmem, stream,
                     h_in, emb, lpz, Wx, wxt, bxv, Wh, wht, bhv, W2, b2, out);
}
